// TPALSTM_TA_7980049236138
// MI455X (gfx1250) — matrix operand order repaired and checked by reading, not compiled, not run on hardware
//
#include <hip/hip_runtime.h>
#include <hip/hip_bf16.h>
#include <stdint.h>

typedef __attribute__((ext_vector_type(16))) _Float16 v16h;
typedef __attribute__((ext_vector_type(8)))  _Float16 v8h;
typedef __attribute__((ext_vector_type(8)))  float    v8f;

namespace {
constexpr int BATCH = 2048;
constexpr int SEQT  = 168;
constexpr int FEAT  = 32;
constexpr int HDIM  = 128;
constexpr int NLAY  = 2;
constexpr int OUTD  = 24;
constexpr int FNUM  = 32;
constexpr int GATE  = 4 * HDIM;      // 512
constexpr int TM1   = SEQT - 1;      // 167
constexpr int CCAT  = HDIM + FNUM;   // 160
constexpr int CDIM  = CCAT + FEAT;   // 192
}

__device__ __forceinline__ v16h join8(v8h lo, v8h hi) {
  union { v16h v; v8h h[2]; } u;
  u.h[0] = lo; u.h[1] = hi;
  return u.v;
}

__device__ __forceinline__ float sigmoidf_(float x) {
  return 1.f / (1.f + __expf(-x));
}

// ---------------------------------------------------------------------------
// f32 -> f16 pack
// ---------------------------------------------------------------------------
__global__ __launch_bounds__(256) void pack_f16_kernel(
    const float* __restrict__ s, _Float16* __restrict__ d, int n) {
  int i = blockIdx.x * 256 + threadIdx.x;
  if (i < n) d[i] = (_Float16)s[i];
}

// ---------------------------------------------------------------------------
// xc = relu(x @ W_hid^T + b_hid)   (B*T,32) @ (32,128) -> f16 (T,B,HD)
// One wave: 16 batch rows x 128 cols for one t (8 WMMA tiles, K=32 = 1 step).
// ---------------------------------------------------------------------------
__global__ __launch_bounds__(256) void proj_kernel(
    const _Float16* __restrict__ x16, const _Float16* __restrict__ Whid16,
    const float* __restrict__ b_hid, _Float16* __restrict__ xh) {
  const int tid  = threadIdx.x;
  const int wave = tid >> 5, lane = tid & 31;
  const int lm = lane & 15, lh = lane >> 4;
  const int t  = blockIdx.x;                 // 0..167
  const int b0 = blockIdx.y * 128 + wave * 16;

  // A fragment (16x32 f16): lane holds row M=lm, K=lh*16..lh*16+15 contiguous
  const _Float16* ar = x16 + ((size_t)(b0 + lm) * SEQT + t) * FEAT;
  v16h a = join8(*(const v8h*)(ar + lh * 16), *(const v8h*)(ar + lh * 16 + 8));

#pragma unroll
  for (int nt = 0; nt < 8; ++nt) {
    const int n = nt * 16 + lm;
    // B fragment (32x16): lane holds col N=lm, K=lh*16..lh*16+15 contiguous
    v16h bf = *(const v16h*)(Whid16 + (size_t)n * FEAT + lh * 16);
    v8f acc = {};
    acc = __builtin_amdgcn_wmma_f32_16x16x32_f16(false, a, false, bf,
                                                 (short)0, acc, false, false);
    const float bb = b_hid[n];
#pragma unroll
    for (int i = 0; i < 8; ++i) {
      float v = acc[i] + bb;
      v = v > 0.f ? v : 0.f;
      xh[((size_t)t * BATCH + (b0 + lh * 8 + i)) * HDIM + n] = (_Float16)v;
    }
  }
}

// ---------------------------------------------------------------------------
// Persistent 2-layer LSTM over T=168 steps. 16 batch rows per workgroup.
// Gates via v_wmma_f32_16x16x32_f16. The layer-0 input slab (4 KB, contiguous
// in the (T,B,HD) layout) is double-buffered in LDS and fetched one step
// ahead with global_load_async_to_lds_b128 (ASYNCcnt), hiding global latency
// behind the previous step's WMMA work. Cell state c lives in registers
// (thread -> fixed (m,n) ownership), h in LDS for WMMA A-fragments.
// LDS: sh 8 KB + stage 8 KB + sg 32 KB + sbias 4 KB = 52 KB.
// ---------------------------------------------------------------------------
__global__ __launch_bounds__(256) void lstm_kernel(
    const _Float16* __restrict__ xh, const _Float16* __restrict__ Wih16,
    const _Float16* __restrict__ Whh16, const float* __restrict__ b_ih,
    const float* __restrict__ b_hh, _Float16* __restrict__ Ht,
    float* __restrict__ htt) {
  __shared__ alignas(16) _Float16 sh[NLAY][16 * HDIM];    // 8 KB
  __shared__ alignas(16) _Float16 stage[2][16 * HDIM];    // 8 KB
  __shared__ float sg[16 * GATE];                         // 32 KB
  __shared__ float sbias[NLAY][GATE];                     // 4 KB

  const int tid  = threadIdx.x;
  const int wave = tid >> 5, lane = tid & 31;
  const int lm = lane & 15, lh = lane >> 4;
  const int b0 = blockIdx.x * 16;

  // per-thread cell state: thread owns (m, n) = (idx>>7, idx&127),
  // idx = tid + r*256, identical mapping every step.
  float creg[NLAY][8];
#pragma unroll
  for (int l = 0; l < NLAY; ++l)
#pragma unroll
    for (int r = 0; r < 8; ++r) creg[l][r] = 0.f;

  for (int i = tid; i < NLAY * GATE; i += 256) {
    int l = i / GATE, n = i % GATE;
    sbias[l][n] = b_ih[l * GATE + n] + b_hh[l * GATE + n];
  }
  for (int i = tid; i < NLAY * 16 * HDIM; i += 256) {
    int l = i / (16 * HDIM), r = i % (16 * HDIM);
    sh[l][r] = (_Float16)0.f;
  }

  // prefetch slab t=0 into stage[0] (16 B per lane, whole slab contiguous)
  {
    const _Float16* gsrc = xh + (size_t)b0 * HDIM + tid * 8;
    const uint32_t ldsa =
        (uint32_t)(uintptr_t)(&stage[0][0]) + (uint32_t)(tid * 16);
    asm volatile("global_load_async_to_lds_b128 %0, %1, off"
                 :: "v"(ldsa), "v"((uint64_t)(uintptr_t)gsrc) : "memory");
  }
  __syncthreads();

  int buf = 0;
  for (int t = 0; t < SEQT; ++t) {
    // slab for step t was issued one step ago; drain + make visible
    asm volatile("s_wait_asynccnt 0x0" ::: "memory");
    __syncthreads();
    // issue slab for step t+1 into the other buffer (its last readers
    // finished before the barrier above)
    if (t + 1 < SEQT) {
      const _Float16* gsrc =
          xh + ((size_t)(t + 1) * BATCH + b0) * HDIM + tid * 8;
      const uint32_t ldsa =
          (uint32_t)(uintptr_t)(&stage[buf ^ 1][0]) + (uint32_t)(tid * 16);
      asm volatile("global_load_async_to_lds_b128 %0, %1, off"
                   :: "v"(ldsa), "v"((uint64_t)(uintptr_t)gsrc) : "memory");
    }

    for (int l = 0; l < NLAY; ++l) {
      // -------- gates = inp @ Wih^T + h @ Whh^T --------
      v16h a_in[4], a_h[4];
      {
        const _Float16* ar =
            (l == 0) ? &stage[buf][lm * HDIM] : &sh[0][lm * HDIM];
#pragma unroll
        for (int kt = 0; kt < 4; ++kt)
          a_in[kt] = join8(*(const v8h*)(ar + kt * 32 + lh * 16),
                           *(const v8h*)(ar + kt * 32 + lh * 16 + 8));
      }
      {
        const _Float16* hr = &sh[l][lm * HDIM];
#pragma unroll
        for (int kt = 0; kt < 4; ++kt)
          a_h[kt] = join8(*(const v8h*)(hr + kt * 32 + lh * 16),
                          *(const v8h*)(hr + kt * 32 + lh * 16 + 8));
      }
      const _Float16* wih = Wih16 + (size_t)l * GATE * HDIM;
      const _Float16* whh = Whh16 + (size_t)l * GATE * HDIM;
#pragma unroll
      for (int j = 0; j < 4; ++j) {
        const int n = (wave * 4 + j) * 16 + lm;   // gate column 0..511
        v8f acc = {};
#pragma unroll
        for (int kt = 0; kt < 4; ++kt) {
          v16h bf = *(const v16h*)(wih + (size_t)n * HDIM + kt * 32 + lh * 16);
          acc = __builtin_amdgcn_wmma_f32_16x16x32_f16(false, a_in[kt], false,
                                                       bf, (short)0, acc,
                                                       false, false);
        }
#pragma unroll
        for (int kt = 0; kt < 4; ++kt) {
          v16h bf = *(const v16h*)(whh + (size_t)n * HDIM + kt * 32 + lh * 16);
          acc = __builtin_amdgcn_wmma_f32_16x16x32_f16(false, a_h[kt], false,
                                                       bf, (short)0, acc,
                                                       false, false);
        }
#pragma unroll
        for (int i = 0; i < 8; ++i) sg[(lh * 8 + i) * GATE + n] = acc[i];
      }
      __syncthreads();

      // -------- elementwise cell update --------
#pragma unroll
      for (int r = 0; r < 8; ++r) {
        const int idx = tid + r * 256;            // 0..2047
        const int m = idx >> 7, n = idx & (HDIM - 1);
        const float gi = sg[m * GATE + n]            + sbias[l][n];
        const float gf = sg[m * GATE + n + HDIM]     + sbias[l][n + HDIM];
        const float gg = sg[m * GATE + n + 2 * HDIM] + sbias[l][n + 2 * HDIM];
        const float go = sg[m * GATE + n + 3 * HDIM] + sbias[l][n + 3 * HDIM];
        const float cn = sigmoidf_(gf) * creg[l][r]
                       + sigmoidf_(gi) * tanhf(gg);
        const float hn = sigmoidf_(go) * tanhf(cn);
        creg[l][r] = cn;
        sh[l][m * HDIM + n] = (_Float16)hn;
        if (l == NLAY - 1) {
          if (t < TM1)
            Ht[((size_t)(b0 + m) * HDIM + n) * TM1 + t] =
                (_Float16)(hn > 0.f ? hn : 0.f);
          else
            htt[(size_t)(b0 + m) * HDIM + n] = hn;
        }
      }
      __syncthreads();
    }
    buf ^= 1;
  }
}

// ---------------------------------------------------------------------------
// conv_vecs / gating attention / concat  (one block per batch row)
// ---------------------------------------------------------------------------
__global__ __launch_bounds__(256) void convattn_kernel(
    const _Float16* __restrict__ Ht, const float* __restrict__ htt,
    const float* __restrict__ conv_k, const float* __restrict__ conv_b,
    const float* __restrict__ W_l1, const float* __restrict__ b_l1,
    float* __restrict__ concat) {
  __shared__ float scv[HDIM * FNUM];   // 16 KB
  __shared__ float shtt[HDIM];
  __shared__ float sw[FNUM];
  __shared__ float salpha[HDIM];
  const int tid = threadIdx.x;
  const int b = blockIdx.x;

  if (tid < HDIM) shtt[tid] = htt[(size_t)b * HDIM + tid];

#pragma unroll
  for (int r = 0; r < 16; ++r) {
    const int idx = tid + r * 256;       // 0..4095
    const int j = idx >> 5, f = idx & 31;
    const _Float16* hp = Ht + ((size_t)b * HDIM + j) * TM1;
    const float* kp = conv_k + (size_t)f * TM1;
    float s = conv_b[f];
    for (int tt = 0; tt < TM1; ++tt) s += (float)hp[tt] * kp[tt];
    scv[j * FNUM + f] = s > 0.f ? s : 0.f;
  }
  __syncthreads();

  if (tid < FNUM) {
    float s = b_l1[tid];
    const float* wp = W_l1 + (size_t)tid * HDIM;
    for (int k = 0; k < HDIM; ++k) s += shtt[k] * wp[k];
    sw[tid] = s;
  }
  __syncthreads();

  if (tid < HDIM) {
    float s = 0.f;
    for (int f = 0; f < FNUM; ++f) s += scv[tid * FNUM + f] * sw[f];
    salpha[tid] = sigmoidf_(s);
  }
  __syncthreads();

  if (tid < HDIM) {
    concat[(size_t)b * CCAT + tid] = shtt[tid];
  } else if (tid < HDIM + FNUM) {
    const int f = tid - HDIM;
    float s = 0.f;
    for (int j = 0; j < HDIM; ++j) s += salpha[j] * scv[j * FNUM + f];
    concat[(size_t)b * CCAT + HDIM + f] = s;
  }
}

// ---------------------------------------------------------------------------
// temporal attention + output head (one block per batch row).
// K=160 (concat) term hoisted out of the T loop.
// ---------------------------------------------------------------------------
__global__ __launch_bounds__(256) void ta_out_kernel(
    const float* __restrict__ x, const float* __restrict__ concat,
    const float* __restrict__ W_ta1, const float* __restrict__ b_ta1,
    const float* __restrict__ W_ta2, const float* __restrict__ W_out,
    const float* __restrict__ b_out, float* __restrict__ out) {
  __shared__ float sconcat[CCAT];
  __shared__ float ss1[SEQT];
  __shared__ float se[SEQT];
  __shared__ float sct[FEAT];
  __shared__ float sinv;
  const int tid = threadIdx.x;
  const int b = blockIdx.x;

  if (tid < CCAT) sconcat[tid] = concat[(size_t)b * CCAT + tid];
  __syncthreads();

  if (tid < SEQT) {
    const float* wp = W_ta1 + (size_t)tid * CDIM;
    float s = b_ta1[tid];
    for (int c = 0; c < CCAT; ++c) s += wp[c] * sconcat[c];
    ss1[tid] = s;
  }
  __syncthreads();

  if (tid < SEQT) {
    float xr[FEAT];
    const float* xp = x + ((size_t)b * SEQT + tid) * FEAT;
#pragma unroll
    for (int c = 0; c < FEAT; ++c) xr[c] = xp[c];
    float e = 0.f;
    for (int o = 0; o < SEQT; ++o) {
      const float* wp = W_ta1 + (size_t)o * CDIM + CCAT;
      float s = ss1[o];
#pragma unroll
      for (int c = 0; c < FEAT; ++c) s += wp[c] * xr[c];
      e += W_ta2[o] * tanhf(s);
    }
    se[tid] = e;
  }
  __syncthreads();

  if (tid == 0) {
    float mx = se[0];
    for (int tt = 1; tt < SEQT; ++tt) mx = se[tt] > mx ? se[tt] : mx;
    float s = 0.f;
    for (int tt = 0; tt < SEQT; ++tt) {
      float v = __expf(se[tt] - mx);
      se[tt] = v;
      s += v;
    }
    sinv = 1.f / s;
  }
  __syncthreads();

  if (tid < FEAT) {
    float s = 0.f;
    for (int tt = 0; tt < SEQT; ++tt)
      s += se[tt] * x[((size_t)b * SEQT + tt) * FEAT + tid];
    sct[tid] = s * sinv;
  }
  __syncthreads();

  if (tid < OUTD) {
    const float* wp = W_out + (size_t)tid * CDIM;
    float s = b_out[tid];
    for (int c = 0; c < CCAT; ++c) s += wp[c] * sconcat[c];
    for (int f = 0; f < FEAT; ++f) s += wp[CCAT + f] * sct[f];
    out[(size_t)b * OUTD + tid] = s;
  }
}

// ---------------------------------------------------------------------------
extern "C" void kernel_launch(void* const* d_in, const int* in_sizes, int n_in,
                              void* d_out, int out_size, void* d_ws,
                              size_t ws_size, hipStream_t stream) {
  (void)in_sizes; (void)n_in; (void)out_size; (void)ws_size;
  const float* x      = (const float*)d_in[0];
  const float* W_hid  = (const float*)d_in[1];
  const float* b_hid  = (const float*)d_in[2];
  const float* W_ih   = (const float*)d_in[3];
  const float* W_hh   = (const float*)d_in[4];
  const float* b_ih   = (const float*)d_in[5];
  const float* b_hh   = (const float*)d_in[6];
  const float* conv_k = (const float*)d_in[7];
  const float* conv_b = (const float*)d_in[8];
  const float* W_l1   = (const float*)d_in[9];
  const float* b_l1   = (const float*)d_in[10];
  const float* W_ta1  = (const float*)d_in[11];
  const float* b_ta1  = (const float*)d_in[12];
  const float* W_ta2  = (const float*)d_in[13];
  const float* W_out  = (const float*)d_in[14];
  const float* b_out  = (const float*)d_in[15];

  char* ws = (char*)d_ws;
  size_t off = 0;
  auto carve = [&](size_t bytes) -> char* {
    char* p = ws + off;
    off += (bytes + 255) & ~(size_t)255;
    return p;
  };
  constexpr size_t n_x16  = (size_t)BATCH * SEQT * FEAT;
  constexpr size_t n_whid = (size_t)HDIM * FEAT;
  constexpr size_t n_wih  = (size_t)NLAY * GATE * HDIM;
  constexpr size_t n_xh   = (size_t)SEQT * BATCH * HDIM;
  constexpr size_t n_Ht   = (size_t)BATCH * HDIM * TM1;

  _Float16* x16    = (_Float16*)carve(n_x16 * 2);
  _Float16* Whid16 = (_Float16*)carve(n_whid * 2);
  _Float16* Wih16  = (_Float16*)carve(n_wih * 2);
  _Float16* Whh16  = (_Float16*)carve(n_wih * 2);
  _Float16* xh     = (_Float16*)carve(n_xh * 2);
  _Float16* Ht     = (_Float16*)carve(n_Ht * 2);
  float*    htt    = (float*)carve((size_t)BATCH * HDIM * 4);
  float*    concat = (float*)carve((size_t)BATCH * CCAT * 4);

  pack_f16_kernel<<<(n_x16 + 255) / 256, 256, 0, stream>>>(x, x16, (int)n_x16);
  pack_f16_kernel<<<(n_whid + 255) / 256, 256, 0, stream>>>(W_hid, Whid16,
                                                            (int)n_whid);
  pack_f16_kernel<<<(n_wih + 255) / 256, 256, 0, stream>>>(W_ih, Wih16,
                                                           (int)n_wih);
  pack_f16_kernel<<<(n_wih + 255) / 256, 256, 0, stream>>>(W_hh, Whh16,
                                                           (int)n_wih);

  proj_kernel<<<dim3(SEQT, BATCH / 128), 256, 0, stream>>>(x16, Whid16, b_hid,
                                                           xh);
  lstm_kernel<<<BATCH / 16, 256, 0, stream>>>(xh, Wih16, Whh16, b_ih, b_hh, Ht,
                                              htt);
  convattn_kernel<<<BATCH, 256, 0, stream>>>(Ht, htt, conv_k, conv_b, W_l1,
                                             b_l1, concat);
  ta_out_kernel<<<BATCH, 256, 0, stream>>>(x, concat, W_ta1, b_ta1, W_ta2,
                                           W_out, b_out, (float*)d_out);
}
